// pairnormRGCN_51118700757094
// MI455X (gfx1250) — compile-verified, matrix operand order checked by reading, not run on hardware
//
#include <hip/hip_runtime.h>
#include <hip/hip_bf16.h>

typedef __bf16 bf16_t;
typedef __attribute__((ext_vector_type(16))) __bf16 v16bf;
typedef __attribute__((ext_vector_type(8)))  __bf16 v8bf;
typedef __attribute__((ext_vector_type(4)))  __bf16 v4bf;
typedef __attribute__((ext_vector_type(8)))  float  v8f;

#define NUM_REL   5
#define NUM_BASES 8
#define HID       128
#define OUTF      64
#define IN_FEAT   256
#define N_PER_TYPE 50000
#define N_NODES   (2 * N_PER_TYPE)
#define N_EDGES   1000000
#define PN_EPS    1e-5f

static __device__ __forceinline__ void atomicAddF(float* p, float v) {
    __hip_atomic_fetch_add(p, v, __ATOMIC_RELAXED, __HIP_MEMORY_SCOPE_AGENT);
}

// ---------------------------------------------------------------------------
// Tiled GEMM: fp32 A x bf16 B^T -> fp32 C, fp32 accumulation in WMMA.
//   - Block tile 128(M) x 64(N), 256 threads = 8 wave32 waves.
//   - Each wave owns a 32x32 quadrant = 2x2 WMMA 16x16 tiles (4 wmma / K-step).
//   - K stepped by 32; LDS double-buffered; global loads staged through
//     registers so HBM latency overlaps the WMMAs; one barrier per K-step.
//   - B is stored pre-transposed (BT[N][K]) so both LDS stages are fully
//     vectorized (b128 loads, b64 stores) with zero bank-transpose work.
//   - M guard via index clamp (garbage rows only feed unstored C rows).
// ---------------------------------------------------------------------------
template<bool ADD, bool RELU, bool BIAS>
__global__ __launch_bounds__(256)
void gemm_wmma_bf16(const float* __restrict__ A, const bf16_t* __restrict__ BT,
                    const float* __restrict__ bias, float* __restrict__ C,
                    int M, int N, int K)
{
    __shared__ __align__(16) bf16_t lgA[2][128][32];  // [buf][m][k]
    __shared__ __align__(16) bf16_t lgB[2][64][32];   // [buf][n][k]

    const int tid   = threadIdx.x;
    const int lane  = tid & 31;
    const int wave  = tid >> 5;
    const int wm    = wave & 3;    // 32-row quadrant
    const int wn    = wave >> 2;   // 32-col quadrant
    const long mBase = (long)blockIdx.x * 128;
    const int  nBase = blockIdx.y * 64;

    const int row16 = lane & 15;
    const int kb    = (lane >> 4) * 8;   // lane-half k base (0 or 8)

    v8f acc[2][2] = {};
    const int nTiles = K >> 5;

    float4 aReg[4];
    v4bf   bReg[2];

    // ---- global -> registers for K-tile kt ----
    auto loadTile = [&](int kt) {
        const int k0 = kt << 5;
        #pragma unroll
        for (int j = 0; j < 4; ++j) {
            int idx = tid + j * 256;          // 1024 float4 = 128x32 fp32
            int r = idx >> 3, c4 = idx & 7;
            long gr = mBase + r; if (gr > (long)M - 1) gr = (long)M - 1;
            aReg[j] = *(const float4*)&A[gr * K + k0 + c4 * 4];
        }
        #pragma unroll
        for (int j = 0; j < 2; ++j) {
            int idx = tid + j * 256;          // 512 v4bf = 64x32 bf16
            int nn = idx >> 3, c4 = idx & 7;
            bReg[j] = *(const v4bf*)&BT[(long)(nBase + nn) * K + k0 + c4 * 4];
        }
    };
    // ---- registers -> LDS buffer ----
    auto stageTile = [&](int buf) {
        #pragma unroll
        for (int j = 0; j < 4; ++j) {
            int idx = tid + j * 256;
            int r = idx >> 3, c4 = idx & 7;
            v4bf t;
            t[0] = (bf16_t)aReg[j].x; t[1] = (bf16_t)aReg[j].y;
            t[2] = (bf16_t)aReg[j].z; t[3] = (bf16_t)aReg[j].w;
            *(v4bf*)&lgA[buf][r][c4 * 4] = t;
        }
        #pragma unroll
        for (int j = 0; j < 2; ++j) {
            int idx = tid + j * 256;
            int nn = idx >> 3, c4 = idx & 7;
            *(v4bf*)&lgB[buf][nn][c4 * 4] = bReg[j];
        }
    };

    loadTile(0);
    stageTile(0);
    __syncthreads();

    for (int kt = 0; kt < nTiles; ++kt) {
        const int buf = kt & 1;
        const bool more = (kt + 1) < nTiles;
        if (more) loadTile(kt + 1);   // HBM loads in flight during WMMAs

        // Fragments per ISA 16-bit layout: half 0 k{0..7,16..23}, half 1 k{8..15,24..31}
        v16bf av[2], bv[2];
        #pragma unroll
        for (int mi = 0; mi < 2; ++mi) {
            const bf16_t* p = &lgA[buf][32 * wm + 16 * mi + row16][kb];
            v8bf lo = *(const v8bf*)p;
            v8bf hi = *(const v8bf*)(p + 16);
            av[mi] = __builtin_shufflevector(lo, hi, 0,1,2,3,4,5,6,7,8,9,10,11,12,13,14,15);
        }
        #pragma unroll
        for (int ni = 0; ni < 2; ++ni) {
            const bf16_t* p = &lgB[buf][32 * wn + 16 * ni + row16][kb];
            v8bf lo = *(const v8bf*)p;
            v8bf hi = *(const v8bf*)(p + 16);
            bv[ni] = __builtin_shufflevector(lo, hi, 0,1,2,3,4,5,6,7,8,9,10,11,12,13,14,15);
        }
        #pragma unroll
        for (int mi = 0; mi < 2; ++mi)
            #pragma unroll
            for (int ni = 0; ni < 2; ++ni)
                acc[mi][ni] = __builtin_amdgcn_wmma_f32_16x16x32_bf16(
                    false, av[mi], false, bv[ni], (short)0, acc[mi][ni], false, false);

        if (more) stageTile(buf ^ 1);
        __syncthreads();
    }

    // ---- store: C/D layout: VGPR v -> M = v + 8*(lane>=16), N = lane&15 ----
    #pragma unroll
    for (int mi = 0; mi < 2; ++mi) {
        const long rbase = mBase + 32 * wm + 16 * mi + ((lane >> 4) << 3);
        #pragma unroll
        for (int ni = 0; ni < 2; ++ni) {
            const int col = nBase + 32 * wn + 16 * ni + (lane & 15);
            #pragma unroll
            for (int v = 0; v < 8; ++v) {
                long gr = rbase + v;
                if (gr < M) {
                    long idx = gr * N + col;
                    float val = acc[mi][ni][v];
                    if (BIAS) val += bias[col];
                    if (ADD)  val += C[idx];
                    if (RELU) val = fmaxf(val, 0.0f);
                    C[idx] = val;
                }
            }
        }
    }
}

// ---------------------------------------------------------------------------
// Small helper kernels
// ---------------------------------------------------------------------------
__global__ __launch_bounds__(256)
void zero_f32(float* __restrict__ p, long n) {
    long i  = (long)blockIdx.x * blockDim.x + threadIdx.x;
    long st = (long)gridDim.x * blockDim.x;
    for (; i < n; i += st) p[i] = 0.0f;
}

// fp32 [K,N] row-major -> bf16 transposed [N,K]
__global__ __launch_bounds__(256)
void cvt_bf16_T(const float* __restrict__ in, bf16_t* __restrict__ outT,
                int K, int N) {
    int i = blockIdx.x * blockDim.x + threadIdx.x;
    if (i >= K * N) return;
    int k = i / N, n = i % N;
    outT[(long)n * K + k] = (bf16_t)in[i];
}

// WcT[fo][r*Fi+fi] = sum_b comp[r,b] * basis[b,fi,fo]   (bf16, transposed)
__global__ __launch_bounds__(256)
void combine_basis_T(const float* __restrict__ comp, const float* __restrict__ basis,
                     bf16_t* __restrict__ WT, int R, int NB, int Fi, int Fo) {
    int i = blockIdx.x * blockDim.x + threadIdx.x;
    int tot = R * Fi * Fo;
    if (i >= tot) return;
    int r   = i / (Fi * Fo);
    int rem = i % (Fi * Fo);
    int fi  = rem / Fo;
    int fo  = rem % Fo;
    float s = 0.0f;
    #pragma unroll
    for (int b = 0; b < NUM_BASES; ++b)
        s += comp[r * NB + b] * basis[((long)b * Fi + fi) * Fo + fo];
    WT[(long)fo * (R * Fi) + (r * Fi + fi)] = (bf16_t)s;
}

// One wave per edge (4 edges / 128-thread block); float4 reads, f32 atomics.
__global__ __launch_bounds__(128)
void scatter_edges(const float* __restrict__ x, const int* __restrict__ ei,
                   const int* __restrict__ et, float* __restrict__ agg,
                   float* __restrict__ cnt, int E, int R) {
    int e = blockIdx.x * 4 + (threadIdx.x >> 5);
    if (e >= E) return;
    int lane = threadIdx.x & 31;
    int s = ei[e];
    int d = ei[E + e];
    int t = et[e];
    const float4* xs = (const float4*)(x + (long)s * HID);   // 32 float4 = 128 f
    float* a = agg + ((long)d * R + t) * HID;
    float4 v = xs[lane];
    atomicAddF(&a[lane * 4 + 0], v.x);
    atomicAddF(&a[lane * 4 + 1], v.y);
    atomicAddF(&a[lane * 4 + 2], v.z);
    atomicAddF(&a[lane * 4 + 3], v.w);
    if (lane == 0)
        atomicAddF(&cnt[(long)d * R + t], 1.0f);
}

__global__ __launch_bounds__(256)
void mean_div(float* __restrict__ agg, const float* __restrict__ cnt, long n, int F) {
    long i  = (long)blockIdx.x * blockDim.x + threadIdx.x;
    long st = (long)gridDim.x * blockDim.x;
    for (; i < n; i += st) {
        long seg = i / F;
        agg[i] = agg[i] / fmaxf(cnt[seg], 1.0f);
    }
}

// PairNorm pass 1: stats[0..F-1] += colsum, stats[F] += sum(x^2)
template<int F>
__global__ __launch_bounds__(F)
void pn_reduce(const float* __restrict__ x, float* __restrict__ stats,
               int Nn, int rowsPerBlock) {
    __shared__ float sh[F];
    int f = threadIdx.x;
    long r0 = (long)blockIdx.x * rowsPerBlock;
    long r1 = r0 + rowsPerBlock; if (r1 > Nn) r1 = Nn;
    float s = 0.0f, ss = 0.0f;
    for (long r = r0; r < r1; ++r) {
        float v = x[r * F + f];
        s  += v;
        ss += v * v;
    }
    atomicAddF(&stats[f], s);
    sh[f] = ss;
    __syncthreads();
    for (int o = F / 2; o > 0; o >>= 1) {
        if (f < o) sh[f] += sh[f + o];
        __syncthreads();
    }
    if (f == 0) atomicAddF(&stats[F], sh[0]);
}

// PairNorm pass 2: stats[f] <- colmean, stats[F+1] <- 1/sqrt(eps + var)
template<int F>
__global__ __launch_bounds__(F)
void pn_finalize(float* __restrict__ stats, int Nn) {
    __shared__ float sh[F];
    int f = threadIdx.x;
    float m = stats[f] / (float)Nn;
    sh[f] = m * m;
    __syncthreads();
    for (int o = F / 2; o > 0; o >>= 1) {
        if (f < o) sh[f] += sh[f + o];
        __syncthreads();
    }
    if (f == 0) {
        float meansq = stats[F] / (float)Nn - sh[0];
        stats[F + 1] = rsqrtf(PN_EPS + meansq);
    }
    stats[f] = m;
}

__global__ __launch_bounds__(256)
void pn_scale(const float* __restrict__ x, const float* __restrict__ stats,
              float* __restrict__ out, long n, int F) {
    long i  = (long)blockIdx.x * blockDim.x + threadIdx.x;
    long st = (long)gridDim.x * blockDim.x;
    float inv = stats[F + 1];
    for (; i < n; i += st)
        out[i] = (x[i] - stats[i % F]) * inv;
}

// ---------------------------------------------------------------------------
// Host orchestration
// ---------------------------------------------------------------------------
static inline char* bump(char*& cur, size_t bytes) {
    char* p = cur;
    cur += (bytes + 255) & ~(size_t)255;
    return p;
}

extern "C" void kernel_launch(void* const* d_in, const int* in_sizes, int n_in,
                              void* d_out, int out_size, void* d_ws, size_t ws_size,
                              hipStream_t stream) {
    (void)in_sizes; (void)n_in; (void)out_size; (void)ws_size;

    const float* x_author = (const float*)d_in[0];
    const float* x_paper  = (const float*)d_in[1];
    const int*   ei       = (const int*)d_in[2];
    const int*   et       = (const int*)d_in[3];
    const float* pa_w     = (const float*)d_in[4];
    const float* pa_b     = (const float*)d_in[5];
    const float* pp_w     = (const float*)d_in[6];
    const float* pp_b     = (const float*)d_in[7];
    const float* comp0    = (const float*)d_in[8];
    const float* basis0   = (const float*)d_in[9];
    const float* root0    = (const float*)d_in[10];
    const float* comp1    = (const float*)d_in[11];
    const float* basis1   = (const float*)d_in[12];
    const float* root1    = (const float*)d_in[13];

    float* out_x      = (float*)d_out;                 // [N, 64]
    float* out_latent = out_x + (long)N_NODES * OUTF;  // [N, 128]

    // workspace carve
    char* cur = (char*)d_ws;
    float*  x0    = (float*)bump(cur, (size_t)N_NODES * HID * 4);
    float*  agg   = (float*)bump(cur, (size_t)N_NODES * NUM_REL * HID * 4);
    float*  cnt   = (float*)bump(cur, (size_t)N_NODES * NUM_REL * 4);
    float*  h1    = (float*)bump(cur, (size_t)N_NODES * HID * 4);
    float*  stats = (float*)bump(cur, 1024);
    bf16_t* WpaT  = (bf16_t*)bump(cur, (size_t)IN_FEAT * HID * 2);        // [128,256]
    bf16_t* WppT  = (bf16_t*)bump(cur, (size_t)IN_FEAT * HID * 2);
    bf16_t* Wc0T  = (bf16_t*)bump(cur, (size_t)NUM_REL * HID * HID * 2);  // [128,640]
    bf16_t* R0T   = (bf16_t*)bump(cur, (size_t)HID * HID * 2);            // [128,128]
    bf16_t* Wc1T  = (bf16_t*)bump(cur, (size_t)NUM_REL * HID * OUTF * 2); // [64,640]
    bf16_t* R1T   = (bf16_t*)bump(cur, (size_t)HID * OUTF * 2);           // [64,128]
    float*  c1    = x0;  // reuse: x0 dead before c1 is written

    const long AGG_N = (long)N_NODES * NUM_REL * HID;

    // --- weight prep (all stored transposed: BT[N][K]) ---
    cvt_bf16_T<<<(IN_FEAT * HID + 255) / 256, 256, 0, stream>>>(pa_w, WpaT, IN_FEAT, HID);
    cvt_bf16_T<<<(IN_FEAT * HID + 255) / 256, 256, 0, stream>>>(pp_w, WppT, IN_FEAT, HID);
    cvt_bf16_T<<<(HID * HID + 255) / 256, 256, 0, stream>>>(root0, R0T, HID, HID);
    cvt_bf16_T<<<(HID * OUTF + 255) / 256, 256, 0, stream>>>(root1, R1T, HID, OUTF);
    combine_basis_T<<<(NUM_REL * HID * HID + 255) / 256, 256, 0, stream>>>(
        comp0, basis0, Wc0T, NUM_REL, NUM_BASES, HID, HID);
    combine_basis_T<<<(NUM_REL * HID * OUTF + 255) / 256, 256, 0, stream>>>(
        comp1, basis1, Wc1T, NUM_REL, NUM_BASES, HID, OUTF);

    // --- input projections: x0 = relu(X @ Wp + b) ---
    {
        dim3 g((N_PER_TYPE + 127) / 128, HID / 64);
        gemm_wmma_bf16<false, true, true><<<g, 256, 0, stream>>>(
            x_author, WpaT, pa_b, x0, N_PER_TYPE, HID, IN_FEAT);
        gemm_wmma_bf16<false, true, true><<<g, 256, 0, stream>>>(
            x_paper, WppT, pp_b, x0 + (long)N_PER_TYPE * HID, N_PER_TYPE, HID, IN_FEAT);
    }

    // --- RGCN layer 0 ---
    zero_f32<<<4096, 256, 0, stream>>>(agg, AGG_N);
    zero_f32<<<2048, 256, 0, stream>>>(cnt, (long)N_NODES * NUM_REL);
    scatter_edges<<<(N_EDGES + 3) / 4, 128, 0, stream>>>(x0, ei, et, agg, cnt, N_EDGES, NUM_REL);
    mean_div<<<4096, 256, 0, stream>>>(agg, cnt, AGG_N, HID);
    {
        dim3 g((N_NODES + 127) / 128, HID / 64);
        gemm_wmma_bf16<false, false, false><<<g, 256, 0, stream>>>(
            agg, Wc0T, nullptr, h1, N_NODES, HID, NUM_REL * HID);
        gemm_wmma_bf16<true, true, false><<<g, 256, 0, stream>>>(
            x0, R0T, nullptr, h1, N_NODES, HID, HID);
    }
    // PairNorm -> x_latent (written straight into d_out slot)
    zero_f32<<<1, 256, 0, stream>>>(stats, 130);
    pn_reduce<HID><<<(N_NODES + 127) / 128, HID, 0, stream>>>(h1, stats, N_NODES, 128);
    pn_finalize<HID><<<1, HID, 0, stream>>>(stats, N_NODES);
    pn_scale<<<4096, 256, 0, stream>>>(h1, stats, out_latent, (long)N_NODES * HID, HID);

    // --- RGCN layer 1 (input = x_latent) ---
    zero_f32<<<4096, 256, 0, stream>>>(agg, AGG_N);
    zero_f32<<<2048, 256, 0, stream>>>(cnt, (long)N_NODES * NUM_REL);
    scatter_edges<<<(N_EDGES + 3) / 4, 128, 0, stream>>>(out_latent, ei, et, agg, cnt, N_EDGES, NUM_REL);
    mean_div<<<4096, 256, 0, stream>>>(agg, cnt, AGG_N, HID);
    {
        dim3 g((N_NODES + 127) / 128, OUTF / 64);
        gemm_wmma_bf16<false, false, false><<<g, 256, 0, stream>>>(
            agg, Wc1T, nullptr, c1, N_NODES, OUTF, NUM_REL * HID);
        gemm_wmma_bf16<true, false, false><<<g, 256, 0, stream>>>(
            out_latent, R1T, nullptr, c1, N_NODES, OUTF, HID);
    }
    // PairNorm -> final x
    zero_f32<<<1, 256, 0, stream>>>(stats, 130);
    pn_reduce<OUTF><<<(N_NODES + 127) / 128, OUTF, 0, stream>>>(c1, stats, N_NODES, 128);
    pn_finalize<OUTF><<<1, OUTF, 0, stream>>>(stats, N_NODES);
    pn_scale<<<4096, 256, 0, stream>>>(c1, stats, out_x, (long)N_NODES * OUTF, OUTF);
}
